// Attention_54683523613054
// MI455X (gfx1250) — matrix operand order repaired and checked by reading, not compiled, not run on hardware
//
#include <hip/hip_runtime.h>
#include <hip/hip_bf16.h>
#include <math.h>

#define B_  4
#define S_  2048
#define D_  1024
#define H_  16
#define DH_ 64
#define OD_ (H_*DH_)

typedef __attribute__((ext_vector_type(16))) __bf16 v16bf;
typedef __attribute__((ext_vector_type(8)))  __bf16 bf8v;
typedef __attribute__((ext_vector_type(8)))  float  v8f;

static __device__ __forceinline__ v16bf cat8(bf8v lo, bf8v hi) {
    return __builtin_shufflevector(lo, hi, 0,1,2,3,4,5,6,7,8,9,10,11,12,13,14,15);
}

// Async copy 16B from global to LDS (gfx1250 GLOBAL_LOAD_ASYNC_TO_LDS_B128,
// tracked by ASYNCcnt). LDS offset = low 32 bits of the generic pointer.
#define ASYNC_LDS_B128(ldsptr, gptr)                                        \
    asm volatile("global_load_async_to_lds_b128 %0, %1, off"                \
                 :: "v"((unsigned)(uintptr_t)(ldsptr)), "v"(gptr) : "memory")

// ---------------------------------------------------------------------------
// Kernel 1: QKV projection. X[8192,1024]fp32 @ W[1024,1024]fp32 -> bf16.
// grid = (M/128, OD/64, 3), block = 128 (4 waves). Tile 128x64, BK=64.
// Each wave owns two 16-row M-tiles -> 16 WMMAs per stage, B-frag reuse 2x.
// Q,V stored [b,h,s,dh]; K stored transposed [b,h,dh,s].
// ---------------------------------------------------------------------------
__global__ __launch_bounds__(128) void qkv_proj_kernel(
    const float* __restrict__ q_seq, const float* __restrict__ k_seq,
    const float* __restrict__ v_seq,
    const float* __restrict__ WQ, const float* __restrict__ WK,
    const float* __restrict__ WV,
    __bf16* __restrict__ Qb, __bf16* __restrict__ Kb, __bf16* __restrict__ Vb)
{
    const int which = blockIdx.z;
    const float* __restrict__ X = (which == 0) ? q_seq : (which == 1) ? k_seq : v_seq;
    const float* __restrict__ W = (which == 0) ? WQ    : (which == 1) ? WK    : WV;
    __bf16* __restrict__ O      = (which == 0) ? Qb    : (which == 1) ? Kb    : Vb;

    const int m0 = blockIdx.x * 128;    // row tile within [B*S]
    const int h  = blockIdx.y;          // head == N-tile (DH_ == 64 == BN)
    const int n0 = h * 64;

    __shared__ __bf16 As[128][72];      // 128x64 bf16, padded
    __shared__ __bf16 Bs[64][72];       // 64x64  bf16, padded

    const int tid = threadIdx.x;
    const int ln  = tid & 31;           // lane in wave
    const int wv  = tid >> 5;           // wave 0..3, owns rows wv*32..+31
    const int lr  = ln & 15;            // lane-low  (C-layout N / A-layout M)
    const int lh  = ln >> 4;            // lane-half

    v8f acc[2][4];
    #pragma unroll
    for (int mt = 0; mt < 2; ++mt)
        #pragma unroll
        for (int nt = 0; nt < 4; ++nt)
            #pragma unroll
            for (int i = 0; i < 8; ++i) acc[mt][nt][i] = 0.0f;

    for (int k0 = 0; k0 < D_; k0 += 64) {
        if (k0 + 64 < D_) {
            __builtin_prefetch(X + (size_t)m0 * D_ + k0 + 64, 0, 1);
            __builtin_prefetch(W + (size_t)(k0 + 64) * OD_ + n0, 0, 1);
        }
        // stage A: 128x64 fp32 -> bf16 (2048 float4 groups / 128 threads)
        #pragma unroll
        for (int i = 0; i < 16; ++i) {
            int idx = tid + i * 128;            // 0..2047
            int r   = idx >> 4;                 // 0..127
            int c4  = (idx & 15) << 2;          // 0..60
            float4 a = *(const float4*)(X + (size_t)(m0 + r) * D_ + k0 + c4);
            As[r][c4+0] = (__bf16)a.x; As[r][c4+1] = (__bf16)a.y;
            As[r][c4+2] = (__bf16)a.z; As[r][c4+3] = (__bf16)a.w;
        }
        // stage B: 64x64 fp32 -> bf16
        #pragma unroll
        for (int i = 0; i < 8; ++i) {
            int idx = tid + i * 128;            // 0..1023
            int r   = idx >> 4;                 // 0..63
            int c4  = (idx & 15) << 2;
            float4 w = *(const float4*)(W + (size_t)(k0 + r) * OD_ + n0 + c4);
            Bs[r][c4+0] = (__bf16)w.x; Bs[r][c4+1] = (__bf16)w.y;
            Bs[r][c4+2] = (__bf16)w.z; Bs[r][c4+3] = (__bf16)w.w;
        }
        __syncthreads();

        #pragma unroll
        for (int kc = 0; kc < 2; ++kc) {        // two K=32 WMMA steps
            v16bf af[2];
            #pragma unroll
            for (int mt = 0; mt < 2; ++mt) {
                // A fragment: row = wv*32 + mt*16 + lr ;
                // e -> k = kc*32 + (e>>3)*16 + lh*8 + (e&7)
                const __bf16* ap = &As[wv * 32 + mt * 16 + lr][kc * 32 + lh * 8];
                af[mt] = cat8(*(const bf8v*)(ap), *(const bf8v*)(ap + 16));
            }
            #pragma unroll
            for (int nt = 0; nt < 4; ++nt) {
                // B fragment: n = nt*16 + lr ; e -> k = kc*32 + (e>>3)*16 + lh*8 + (e&7)
                v16bf bfrag;
                #pragma unroll
                for (int e = 0; e < 16; ++e)
                    bfrag[e] = Bs[kc * 32 + (e >> 3) * 16 + lh * 8 + (e & 7)][nt * 16 + lr];
                #pragma unroll
                for (int mt = 0; mt < 2; ++mt) {
                    acc[mt][nt] = __builtin_amdgcn_wmma_f32_16x16x32_bf16(
                        false, af[mt], false, bfrag, (short)0, acc[mt][nt], false, false);
                }
            }
        }
        __syncthreads();
    }

    // C/D layout: M = lh*8 + r (VGPR r), N = lr
    #pragma unroll
    for (int mt = 0; mt < 2; ++mt) {
        #pragma unroll
        for (int nt = 0; nt < 4; ++nt) {
            #pragma unroll
            for (int r = 0; r < 8; ++r) {
                int m  = m0 + wv * 32 + mt * 16 + lh * 8 + r;  // row in [B*S]
                int b  = m / S_;
                int s  = m - b * S_;
                int dh = nt * 16 + lr;
                __bf16 val = (__bf16)acc[mt][nt][r];
                if (which == 1) {
                    // K transposed: [b,h,dh,s]
                    Kb[(((size_t)(b * H_ + h) * DH_) + dh) * S_ + s] = val;
                } else {
                    // Q,V: [b,h,s,dh]
                    O[(((size_t)(b * H_ + h) * S_) + s) * DH_ + dh] = val;
                }
            }
        }
    }
}

// ---------------------------------------------------------------------------
// Kernel 2: flash attention over bf16 Q/K/V. grid = (S/64, H, B), block = 128.
// Each wave owns 16 q-rows; 64-key blocks, double-buffered async K/V staging
// via global_load_async_to_lds_b128 + s_wait_asynccnt. 16 WMMAs per block.
// ---------------------------------------------------------------------------
__global__ __launch_bounds__(128) void attn_kernel(
    const __bf16* __restrict__ Qb, const __bf16* __restrict__ Kb,
    const __bf16* __restrict__ Vb, float* __restrict__ out)
{
    const int q0 = blockIdx.x * 64;
    const int h  = blockIdx.y;
    const int b  = blockIdx.z;
    const size_t base = (size_t)(b * H_ + h) * S_ * DH_;
    const __bf16* __restrict__ Q  = Qb + base;   // [s][dh]
    const __bf16* __restrict__ Kt = Kb + base;   // [dh][s]  (pre-transposed)
    const __bf16* __restrict__ V  = Vb + base;   // [s][dh]

    __shared__ __bf16 Kts[2][64][72];   // dh x 64 keys, padded (stride 144B)
    __shared__ __bf16 Vs [2][64][72];   // 64 keys x dh, padded (stride 144B)
    __shared__ __bf16 Ps [4][16][72];   // per-wave P scratch 16x64, padded

    const int tid = threadIdx.x;
    const int ln  = tid & 31;
    const int wv  = tid >> 5;
    const int lr  = ln & 15;
    const int lh  = ln >> 4;

    // Q A-fragments (held in registers for whole kernel):
    // row = q0 + wv*16 + lr ; e -> k = kc*32 + (e>>3)*16 + lh*8 + (e&7)
    v16bf qfrag[2];
    {
        const __bf16* qrow = Q + (size_t)(q0 + wv * 16 + lr) * DH_;
        #pragma unroll
        for (int kc = 0; kc < 2; ++kc) {
            const __bf16* qp = qrow + kc * 32 + lh * 8;
            qfrag[kc] = cat8(*(const bf8v*)(qp), *(const bf8v*)(qp + 16));
        }
    }

    float mstate[8], lstate[8];
    v8f accO[4];
    #pragma unroll
    for (int r = 0; r < 8; ++r) { mstate[r] = -1.0e30f; lstate[r] = 0.0f; }
    #pragma unroll
    for (int nt = 0; nt < 4; ++nt)
        #pragma unroll
        for (int i = 0; i < 8; ++i) accO[nt][i] = 0.0f;

    const float SCALE = 0.125f;   // 1/sqrt(64)
    const int   NKB   = S_ / 64;  // 32 key blocks of 64 keys

    // async-stage one 64-key K/V block into buffer `buf` (8KB K + 8KB V)
    auto stage_async = [&](int kb, int buf) {
        #pragma unroll
        for (int i = 0; i < 4; ++i) {
            int idx = tid + i * 128;                 // 0..511 16B groups
            int d   = idx >> 3;                      // 0..63 (dh row / key row)
            int c8  = (idx & 7) << 3;                // 0..56
            ASYNC_LDS_B128(&Kts[buf][d][c8], Kt + (size_t)d * S_ + kb * 64 + c8);
            ASYNC_LDS_B128(&Vs[buf][d][c8],  V + (size_t)(kb * 64 + d) * DH_ + c8);
        }
    };

    stage_async(0, 0);

    for (int kb = 0; kb < NKB; ++kb) {
        const int buf = kb & 1;
        asm volatile("s_wait_asynccnt 0x0" ::: "memory");  // my async stores to LDS done
        __syncthreads();                                   // everyone's stores visible
        if (kb + 1 < NKB) stage_async(kb + 1, buf ^ 1);    // prefetch next block

        // scores: S[16q x 64k] via four 16-key N-tiles, contraction over dh=64
        v8f sc[4];
        #pragma unroll
        for (int nt = 0; nt < 4; ++nt)
            #pragma unroll
            for (int i = 0; i < 8; ++i) sc[nt][i] = 0.0f;
        #pragma unroll
        for (int kc = 0; kc < 2; ++kc) {
            #pragma unroll
            for (int nt = 0; nt < 4; ++nt) {
                v16bf kf;
                #pragma unroll
                for (int e = 0; e < 16; ++e)
                    kf[e] = Kts[buf][kc * 32 + (e >> 3) * 16 + lh * 8 + (e & 7)][nt * 16 + lr];
                sc[nt] = __builtin_amdgcn_wmma_f32_16x16x32_bf16(
                             false, qfrag[kc], false, kf, (short)0, sc[nt], false, false);
            }
        }

        // online softmax; each row (M = lh*8+r) lives across one 16-lane half
        #pragma unroll
        for (int r = 0; r < 8; ++r) {
            float a0 = sc[0][r] * SCALE, a1 = sc[1][r] * SCALE;
            float a2 = sc[2][r] * SCALE, a3 = sc[3][r] * SCALE;
            float mx = fmaxf(fmaxf(a0, a1), fmaxf(a2, a3));
            #pragma unroll
            for (int m = 8; m >= 1; m >>= 1) mx = fmaxf(mx, __shfl_xor(mx, m, 32));
            float mnew = fmaxf(mstate[r], mx);
            float corr = __expf(mstate[r] - mnew);
            float e0 = __expf(a0 - mnew);
            float e1 = __expf(a1 - mnew);
            float e2 = __expf(a2 - mnew);
            float e3 = __expf(a3 - mnew);
            float ps = (e0 + e1) + (e2 + e3);
            #pragma unroll
            for (int m = 8; m >= 1; m >>= 1) ps += __shfl_xor(ps, m, 32);
            lstate[r] = lstate[r] * corr + ps;
            mstate[r] = mnew;
            #pragma unroll
            for (int nt = 0; nt < 4; ++nt) accO[nt][r] *= corr;
            // C-layout -> LDS for A-layout reload (row lh*8+r, cols nt*16+lr)
            Ps[wv][lh * 8 + r][lr]      = (__bf16)e0;
            Ps[wv][lh * 8 + r][16 + lr] = (__bf16)e1;
            Ps[wv][lh * 8 + r][32 + lr] = (__bf16)e2;
            Ps[wv][lh * 8 + r][48 + lr] = (__bf16)e3;
        }
        asm volatile("s_wait_dscnt 0" ::: "memory");  // DS in-order per wave

        // P A-fragments: keys 0-31 (ph=0) and 32-63 (ph=1)
        v16bf pfrag[2];
        #pragma unroll
        for (int ph = 0; ph < 2; ++ph) {
            const __bf16* pp = &Ps[wv][lr][ph * 32 + lh * 8];
            pfrag[ph] = cat8(*(const bf8v*)(pp), *(const bf8v*)(pp + 16));
        }

        // O += P[16x64] x V[64x64]
        #pragma unroll
        for (int ph = 0; ph < 2; ++ph) {
            #pragma unroll
            for (int nt = 0; nt < 4; ++nt) {
                v16bf vf;
                #pragma unroll
                for (int e = 0; e < 16; ++e)
                    vf[e] = Vs[buf][ph * 32 + (e >> 3) * 16 + lh * 8 + (e & 7)][nt * 16 + lr];
                accO[nt] = __builtin_amdgcn_wmma_f32_16x16x32_bf16(
                               false, pfrag[ph], false, vf, (short)0, accO[nt], false, false);
            }
        }
    }

    // normalize and write fp32 output [b, q, h*dh]
    #pragma unroll
    for (int r = 0; r < 8; ++r) {
        float rinv = 1.0f / lstate[r];
        int q = q0 + wv * 16 + lh * 8 + r;
        #pragma unroll
        for (int nt = 0; nt < 4; ++nt) {
            out[((size_t)b * S_ + q) * OD_ + h * DH_ + nt * 16 + lr] = accO[nt][r] * rinv;
        }
    }
}

// ---------------------------------------------------------------------------
extern "C" void kernel_launch(void* const* d_in, const int* in_sizes, int n_in,
                              void* d_out, int out_size, void* d_ws, size_t ws_size,
                              hipStream_t stream) {
    const float* q_seq = (const float*)d_in[0];
    const float* k_seq = (const float*)d_in[1];
    const float* v_seq = (const float*)d_in[2];
    const float* WQ    = (const float*)d_in[3];
    const float* WK    = (const float*)d_in[4];
    const float* WV    = (const float*)d_in[5];
    float* out = (float*)d_out;

    const size_t nElem = (size_t)B_ * H_ * S_ * DH_;   // 8.4M bf16 each
    __bf16* Qb = (__bf16*)d_ws;
    __bf16* Kb = Qb + nElem;
    __bf16* Vb = Kb + nElem;

    dim3 g1((B_ * S_) / 128, OD_ / 64, 3);  // (64, 16, 3)
    qkv_proj_kernel<<<g1, 128, 0, stream>>>(q_seq, k_seq, v_seq, WQ, WK, WV, Qb, Kb, Vb);

    dim3 g2(S_ / 64, H_, B_);               // (32, 16, 4)
    attn_kernel<<<g2, 128, 0, stream>>>(Qb, Kb, Vb, out);
}
